// AttentionLayer_79774722555996
// MI455X (gfx1250) — matrix operand order repaired and checked by reading, not compiled, not run on hardware
//
#include <hip/hip_runtime.h>
#include <hip/hip_bf16.h>

typedef __attribute__((ext_vector_type(16))) __bf16 v16bf;
typedef __attribute__((ext_vector_type(8)))  __bf16 v8bf;
typedef __attribute__((ext_vector_type(8)))  float  v8f;

#define NN      100000
#define EE      500000
#define DIM     256
#define LDP     264   // bf16 LDS row pad (stride 528B -> bank stride 4, conflict-free b128)
#define EHP     268   // f32 LDS row pad for Eh tile
#define SCALE_Q 0.17677669529663687f  // 1/sqrt(32)

// ---------------------------------------------------------------- utilities

__global__ __launch_bounds__(256) void zero_f32(float* __restrict__ p, int n) {
    int i = blockIdx.x * 256 + threadIdx.x;
    if (i < n) p[i] = 0.0f;
}

// WT[w][n*256+k] = bf16(W[w][k*256+n])  for w in {Q,K,V,E}
__global__ __launch_bounds__(256) void convert_w(const float* __restrict__ W0,
                                                 const float* __restrict__ W1,
                                                 const float* __restrict__ W2,
                                                 const float* __restrict__ W3,
                                                 __bf16* __restrict__ WT) {
    int idx = blockIdx.x * 256 + threadIdx.x;      // 0 .. 4*65536-1
    int w = idx >> 16;
    int rem = idx & 65535;
    int n = rem >> 8, k = rem & 255;
    const float* W = (w == 0) ? W0 : (w == 1) ? W1 : (w == 2) ? W2 : W3;
    WT[(size_t)w * 65536 + (size_t)n * 256 + k] = (__bf16)W[(size_t)k * 256 + n];
}

// stage one [16 x 256] f32 tile -> LDS bf16 with padded stride
__device__ __forceinline__ void stage_tile_bf16(const float* __restrict__ g,
                                                __bf16* s, int t) {
    int r = t >> 4, j = t & 15;
    const float* p = g + (size_t)r * DIM + j * 16;
    v8bf a0, a1;
#pragma unroll
    for (int i = 0; i < 8; ++i) { a0[i] = (__bf16)p[i]; a1[i] = (__bf16)p[i + 8]; }
    *(v8bf*)(s + r * LDP + j * 16)     = a0;
    *(v8bf*)(s + r * LDP + j * 16 + 8) = a1;
}

// load all 8 A fragments (K = kt*32 .. +31) for this lane from LDS
// A 16x32 bf16, k order used here (B matches): lanes 0-15 (M=lane) elems0-15 = K 0..15
//                      lanes16-31 (M=lane-16) elems0-15 = K16..31
__device__ __forceinline__ void load_afrags(const __bf16* s, int m, int half,
                                            v16bf* afrag) {
#pragma unroll
    for (int kt = 0; kt < 8; ++kt) {
        const __bf16* p0 = s + m * LDP + kt * 32 + half * 16;
        v8bf lo = *(const v8bf*)p0;
        v8bf hi = *(const v8bf*)(p0 + 8);
        v16bf a;
#pragma unroll
        for (int i = 0; i < 8; ++i) { a[i] = lo[i]; a[i + 8] = hi[i]; }
        afrag[kt] = a;
    }
}

// ---------------------------------------------------------------- Q/K/V GEMM
// block: 16 node rows x full 256 cols x 3 weights. 8 waves x 6 tiles.

__global__ __launch_bounds__(256) void qkv_gemm(const float* __restrict__ x,
                                                const __bf16* __restrict__ WT,
                                                float* __restrict__ Q,
                                                float* __restrict__ K,
                                                float* __restrict__ V) {
    __shared__ __bf16 sx[16 * LDP];
    const int t = threadIdx.x;
    const int row0 = blockIdx.x * 16;

    stage_tile_bf16(x + (size_t)row0 * DIM, sx, t);
    __syncthreads();

    const int wave = t >> 5, lane = t & 31;
    const int m = lane & 15, half = lane >> 4;

    v16bf afrag[8];
    load_afrags(sx, m, half, afrag);

#pragma unroll
    for (int i6 = 0; i6 < 6; ++i6) {
        const int tid  = wave + 8 * i6;      // 0..47
        const int wsel = tid >> 4;           // 0=Q 1=K 2=V
        const int nt   = tid & 15;           // n tile
        const __bf16* WTp = WT + (size_t)wsel * 65536;

        v8f acc = {0.f, 0.f, 0.f, 0.f, 0.f, 0.f, 0.f, 0.f};
#pragma unroll
        for (int kt = 0; kt < 8; ++kt) {
            // B 32x16 bf16: lane = N (+16*half in K); 16 contiguous bf16 of W^T row
            const __bf16* bp = WTp + (size_t)(nt * 16 + m) * 256 + kt * 32 + half * 16;
            v16bf b = *(const v16bf*)bp;
            acc = __builtin_amdgcn_wmma_f32_16x16x32_bf16(
                false, afrag[kt], false, b, (short)0, acc, false, false);
        }
        float*  O    = (wsel == 0) ? Q : (wsel == 1) ? K : V;
        float   mult = (wsel == 0) ? SCALE_Q : 1.0f;
#pragma unroll
        for (int r = 0; r < 8; ++r)
            O[(size_t)(row0 + r + 8 * half) * DIM + nt * 16 + m] = acc[r] * mult;
    }
}

// ------------------------------------------------- fused Eh GEMM + edge stage
// block: 16 edges. Phase 1: Eh tile via WMMA -> LDS. Phase 2: score/exp/scatter.

__global__ __launch_bounds__(256) void edge_fused(const float* __restrict__ ea,
                                                  const __bf16* __restrict__ WTe,
                                                  const long long* __restrict__ ei,
                                                  const float* __restrict__ Q,
                                                  const float* __restrict__ Kf,
                                                  const float* __restrict__ Vf,
                                                  float* __restrict__ wV,
                                                  float* __restrict__ Zb) {
    __shared__ __bf16 sx[16 * LDP];
    __shared__ float  seh[16 * EHP];
    const int t = threadIdx.x;
    const int e0 = blockIdx.x * 16;

    stage_tile_bf16(ea + (size_t)e0 * DIM, sx, t);
    __syncthreads();

    const int wave = t >> 5, lane = t & 31;
    const int m = lane & 15, half = lane >> 4;

    v16bf afrag[8];
    load_afrags(sx, m, half, afrag);

#pragma unroll
    for (int i2 = 0; i2 < 2; ++i2) {
        const int nt = wave * 2 + i2;
        v8f acc = {0.f, 0.f, 0.f, 0.f, 0.f, 0.f, 0.f, 0.f};
#pragma unroll
        for (int kt = 0; kt < 8; ++kt) {
            const __bf16* bp = WTe + (size_t)(nt * 16 + m) * 256 + kt * 32 + half * 16;
            v16bf b = *(const v16bf*)bp;
            acc = __builtin_amdgcn_wmma_f32_16x16x32_bf16(
                false, afrag[kt], false, b, (short)0, acc, false, false);
        }
#pragma unroll
        for (int r = 0; r < 8; ++r)
            seh[(r + 8 * half) * EHP + nt * 16 + m] = acc[r];
    }
    __syncthreads();

    // ---- edge phase: 16 threads per edge, 16 channels per thread ----
    const int e = t >> 4, j = t & 15;
    const long long srcn = ei[e0 + e];
    const long long dstn = ei[(size_t)EE + e0 + e];

    const float* kp  = Kf + (size_t)srcn * DIM + j * 16;
    const float* qp  = Q  + (size_t)dstn * DIM + j * 16;   // pre-scaled by 1/sqrt(DH)
    const float* vp  = Vf + (size_t)srcn * DIM + j * 16;
    const float* ehp = &seh[e * EHP + j * 16];

    float part = 0.0f;
    float vv[16];
#pragma unroll
    for (int i = 0; i < 16; ++i) {
        part += kp[i] * qp[i] * ehp[i];
        vv[i] = vp[i];
    }
    // head h = j/2 spans threads j (even) and j|1; reduce across the pair
    float s = part + __shfl_xor(part, 1, 32);
    s = fminf(5.0f, fmaxf(-5.0f, s));
    const float sc = __expf(s);

    float* wdst = wV + (size_t)dstn * DIM + j * 16;
#pragma unroll
    for (int i = 0; i < 16; ++i)
        unsafeAtomicAdd(&wdst[i], vv[i] * sc);     // global_atomic_add_f32, hits L2
    if ((j & 1) == 0)
        unsafeAtomicAdd(&Zb[(size_t)dstn * 8 + (j >> 1)], sc);
}

// ----------------------------------------- h = x + wV/(Z+eps), BN statistics
// 1 thread = 1 channel, 125 nodes per block -> 800 blocks, coalesced rows.

__global__ __launch_bounds__(256) void node_h_stats(const float* __restrict__ x,
                                                    const float* __restrict__ wV,
                                                    const float* __restrict__ Zb,
                                                    float* __restrict__ hout,
                                                    float* __restrict__ sum,
                                                    float* __restrict__ sumsq) {
    const int c = threadIdx.x;
    const int b = blockIdx.x;
    float s = 0.0f, s2 = 0.0f;
    for (int n = 0; n < 125; ++n) {
        const int node = b * 125 + n;
        const size_t i = (size_t)node * DIM + c;
        const float z = Zb[(size_t)node * 8 + (c >> 5)];
        const float h = x[i] + wV[i] / (z + 1e-6f);
        hout[i] = h;
        s += h;
        s2 += h * h;
    }
    unsafeAtomicAdd(&sum[c], s);
    unsafeAtomicAdd(&sumsq[c], s2);
}

__global__ __launch_bounds__(256) void bn_coef(const float* __restrict__ sum,
                                               const float* __restrict__ sumsq,
                                               const float* __restrict__ gamma,
                                               const float* __restrict__ beta,
                                               float* __restrict__ a,
                                               float* __restrict__ b2) {
    const int c = threadIdx.x;
    const float ninv = 1.0f / (float)NN;
    const float mean = sum[c] * ninv;
    const float var  = sumsq[c] * ninv - mean * mean;   // biased, matches reference
    const float g    = gamma[c] * rsqrtf(var + 1e-5f);
    a[c]  = g;
    b2[c] = beta[c] - mean * g;
}

__global__ __launch_bounds__(256) void bn_apply(float* __restrict__ h,
                                                const float* __restrict__ a,
                                                const float* __restrict__ b2) {
    const size_t i = (size_t)blockIdx.x * 256 + threadIdx.x;
    const int c = threadIdx.x;   // row length == 256 == blockDim
    h[i] = h[i] * a[c] + b2[c];
}

// ---------------------------------------------------------------- launcher

extern "C" void kernel_launch(void* const* d_in, const int* in_sizes, int n_in,
                              void* d_out, int out_size, void* d_ws, size_t ws_size,
                              hipStream_t stream) {
    const float*     x     = (const float*)d_in[0];
    const float*     ea    = (const float*)d_in[1];
    const float*     WQ    = (const float*)d_in[2];
    const float*     WK    = (const float*)d_in[3];
    const float*     WE    = (const float*)d_in[4];
    const float*     WV    = (const float*)d_in[5];
    const float*     gamma = (const float*)d_in[6];
    const float*     beta  = (const float*)d_in[7];
    const long long* ei    = (const long long*)d_in[8];
    float* out = (float*)d_out;

    // workspace layout
    char*   w   = (char*)d_ws;
    __bf16* WT  = (__bf16*)w;                       // 4 x 65536 bf16 (Q,K,V,E) = 512 KB
    float*  Qf  = (float*)(w + 524288);             // N*256
    float*  Kf  = Qf  + (size_t)NN * DIM;
    float*  Vf  = Kf  + (size_t)NN * DIM;
    float*  wVb = Vf  + (size_t)NN * DIM;           // N*256  (zeroed)
    float*  Zb  = wVb + (size_t)NN * DIM;           // N*8    (zeroed)
    float*  sum = Zb  + (size_t)NN * 8;             // 256    (zeroed)
    float*  ssq = sum + 256;                        // 256    (zeroed)
    float*  ac  = ssq + 256;                        // 256
    float*  bc  = ac  + 256;                        // 256

    const int nzero = NN * DIM + NN * 8 + 512;      // wVb..ssq contiguous
    zero_f32<<<(nzero + 255) / 256, 256, 0, stream>>>(wVb, nzero);
    convert_w<<<1024, 256, 0, stream>>>(WQ, WK, WV, WE, WT);
    qkv_gemm<<<NN / 16, 256, 0, stream>>>(x, WT, Qf, Kf, Vf);
    edge_fused<<<EE / 16, 256, 0, stream>>>(ea, WT + (size_t)3 * 65536, ei,
                                            Qf, Kf, Vf, wVb, Zb);
    node_h_stats<<<800, 256, 0, stream>>>(x, wVb, Zb, out, sum, ssq);
    bn_coef<<<1, 256, 0, stream>>>(sum, ssq, gamma, beta, ac, bc);
    bn_apply<<<NN * DIM / 256, 256, 0, stream>>>(out, ac, bc);
}
